// Attention_10041633538134
// MI455X (gfx1250) — matrix operand order repaired and checked by reading, not compiled, not run on hardware
//
#include <hip/hip_runtime.h>

#define SRC_LEN 512
#define BATCH   64
#define HID2    2048   // K = 2*HIDDEN
#define UNIT    1024   // N
#define KC      256    // K chunk staged in LDS
#define PK      (KC + 8)   // padded LDS row stride (elements): +16B -> conflict-free
#define NCHUNK  (HID2 / KC)          // 8
#define NF4     ((64 * KC / 4) / 512) // float4s per thread per chunk = 8

typedef __attribute__((ext_vector_type(16))) __bf16      v16bf;
typedef __attribute__((ext_vector_type(8)))  float       v8f;
typedef __attribute__((ext_vector_type(4)))  unsigned int uint4v;

union Frag {
    uint4v q[2];
    v16bf  v;
};

__device__ __forceinline__ unsigned int pack_bf16(float x, float y) {
    unsigned int ux = __float_as_uint(x);
    unsigned int uy = __float_as_uint(y);
    ux += 0x7fffu + ((ux >> 16) & 1u);   // round-to-nearest-even
    uy += 0x7fffu + ((uy >> 16) & 1u);
    return (ux >> 16) | (uy & 0xffff0000u);
}

// ---------------- Kernel 1: Ua_w f32 -> bf16 (once, stays L2-resident) ----
__global__ __launch_bounds__(256) void cvt_ua_kernel(const float* __restrict__ src,
                                                     unsigned int* __restrict__ dst,
                                                     int n4) {
    int i = blockIdx.x * 256 + threadIdx.x;      // float4 index
    if (i < n4) {
        float4 f = ((const float4*)src)[i];
        uint2 p;
        p.x = pack_bf16(f.x, f.y);
        p.y = pack_bf16(f.z, f.w);
        ((uint2*)dst)[i] = p;
    }
}

// ---------------- Kernel 2: proj[b,u] = dec[b,:]·Wa_w[u,:] + Wa_b[u] + Ua_b[u]
__global__ __launch_bounds__(256) void dec_proj_kernel(const float* __restrict__ dec,
                                                       const float* __restrict__ Wa_w,
                                                       const float* __restrict__ Wa_b,
                                                       const float* __restrict__ Ua_b,
                                                       float* __restrict__ proj) {
    __shared__ float s_dec[1024];
    const int b   = blockIdx.x;
    const int tid = threadIdx.x;
    ((float4*)s_dec)[tid] = ((const float4*)(dec + (size_t)b * 1024))[tid];
    __syncthreads();
    #pragma unroll
    for (int uu = 0; uu < 4; ++uu) {
        const int u = tid + uu * 256;
        const float4* w = (const float4*)(Wa_w + (size_t)u * 1024);
        float a0 = 0.f, a1 = 0.f, a2 = 0.f, a3 = 0.f;
        #pragma unroll 4
        for (int k4 = 0; k4 < 256; ++k4) {
            float4 wv = w[k4];
            float4 dv = ((const float4*)s_dec)[k4];
            a0 += wv.x * dv.x; a1 += wv.y * dv.y;
            a2 += wv.z * dv.z; a3 += wv.w * dv.w;
        }
        proj[(size_t)b * UNIT + u] = (a0 + a1) + (a2 + a3) + Wa_b[u] + Ua_b[u];
    }
}

// ---------------- Kernel 3: fused GEMM(bf16 WMMA) + tanh + v-dot ----------
// 1 block per s (64 batch rows x K=2048, contiguous). 16 waves: 4(M) x 4(N).
// A chunks are register double-buffered: loads for chunk c+1 are issued before
// the 128-WMMA compute block of chunk c and only consumed after the next
// barrier, hiding HBM latency under WMMA issue.
__global__ __launch_bounds__(512) void attn_main_kernel(
        const float*          __restrict__ enc,     // [512*64, 2048]
        const unsigned short* __restrict__ ua_bf,   // [1024, 2048] bf16
        const float*          __restrict__ proj,    // [64, 1024]
        const float*          __restrict__ v_w,     // [1024]
        float*                __restrict__ logits)  // [64, 512]
{
    __shared__ unsigned short As[64 * PK];          // 33.8 KB, bank-pad +8 elems/row
    __shared__ float s_logit[BATCH];

    const int tid   = threadIdx.x;
    const int s     = blockIdx.x;
    const int lane  = tid & 31;
    const int wid   = tid >> 5;
    const int waveM = wid & 3;       // row tile:  rows 16*waveM .. +15
    const int waveN = wid >> 2;      // col span:  cols 256*waveN .. +255
    const int l16   = lane & 15;
    const int hh    = lane >> 4;

    if (tid < BATCH) s_logit[tid] = 0.f;

    v8f acc[16];
    #pragma unroll
    for (int i = 0; i < 16; ++i) { v8f z = {}; acc[i] = z; }

    const float* encBase = enc + (size_t)s * BATCH * HID2;
    const int arow = waveM * 16 + l16;

    // per-thread staging geometry (constant across chunks)
    const int flatBase = tid;                 // float4 index stride 512 per it
    // preload chunk 0 into registers
    float4 st[NF4];
    #pragma unroll
    for (int it = 0; it < NF4; ++it) {
        const int flat = flatBase + it * 512;
        const int r    = flat / (KC / 4);
        const int c4   = flat % (KC / 4);
        st[it] = *(const float4*)(encBase + (size_t)r * HID2 + 0 + c4 * 4);
    }

    for (int c = 0; c < NCHUNK; ++c) {
        const int kc = c * KC;
        __syncthreads();                       // LDS free (prev chunk readers done)
        // convert held registers -> bf16 LDS
        #pragma unroll
        for (int it = 0; it < NF4; ++it) {
            const int flat = flatBase + it * 512;
            const int r    = flat / (KC / 4);
            const int c4   = flat % (KC / 4);
            unsigned int* dst = (unsigned int*)&As[r * PK + c4 * 4];
            dst[0] = pack_bf16(st[it].x, st[it].y);
            dst[1] = pack_bf16(st[it].z, st[it].w);
        }
        __syncthreads();                       // LDS ready
        // issue next chunk's global loads; consumed only after next barrier
        if (c + 1 < NCHUNK) {
            #pragma unroll
            for (int it = 0; it < NF4; ++it) {
                const int flat = flatBase + it * 512;
                const int r    = flat / (KC / 4);
                const int c4   = flat % (KC / 4);
                st[it] = *(const float4*)(encBase + (size_t)r * HID2 + (kc + KC) + c4 * 4);
            }
        }

        #pragma unroll
        for (int kk = 0; kk < KC; kk += 32) {
            // A fragment: lane row = arow, K = {kk+8h..+7, kk+16+8h..+7}
            Frag a;
            a.q[0] = *(const uint4v*)&As[arow * PK + kk + hh * 8];
            a.q[1] = *(const uint4v*)&As[arow * PK + kk + 16 + hh * 8];
            // B fragments: lane col = n, K = {kk+8h..+7, kk+16+8h..+7}
            const unsigned short* bcol =
                ua_bf + (size_t)(waveN * 256 + l16) * HID2 + (kc + kk) + hh * 8;
            #pragma unroll
            for (int nt = 0; nt < 16; ++nt) {
                Frag b;
                b.q[0] = *(const uint4v*)(bcol + (size_t)nt * 16 * HID2);
                b.q[1] = *(const uint4v*)(bcol + (size_t)nt * 16 * HID2 + 16);
                acc[nt] = __builtin_amdgcn_wmma_f32_16x16x32_bf16(
                    false, a.v, false, b.v, (short)0, acc[nt], false, false);
            }
        }
    }

    // epilogue: e = tanh(C + proj) * v; reduce over u
    float rowsum[8];
    #pragma unroll
    for (int i = 0; i < 8; ++i) rowsum[i] = 0.f;

    #pragma unroll
    for (int nt = 0; nt < 16; ++nt) {
        const int   u  = waveN * 256 + nt * 16 + l16;
        const float vw = v_w[u];
        #pragma unroll
        for (int i = 0; i < 8; ++i) {
            const int b = waveM * 16 + hh * 8 + i;   // C: VGPR i -> M = i + 8*half
            const float e = tanhf(acc[nt][i] + proj[(size_t)b * UNIT + u]);
            rowsum[i] += e * vw;
        }
    }
    #pragma unroll
    for (int i = 0; i < 8; ++i) {
        float v = rowsum[i];
        v += __shfl_xor(v, 1, 32);
        v += __shfl_xor(v, 2, 32);
        v += __shfl_xor(v, 4, 32);
        v += __shfl_xor(v, 8, 32);
        if (l16 == 0) {
            const int b = waveM * 16 + hh * 8 + i;
            atomicAdd(&s_logit[b], v);               // ds_add_f32
        }
    }
    __syncthreads();
    if (tid < BATCH) logits[(size_t)tid * SRC_LEN + s] = s_logit[tid];
}

// ---------------- Kernel 4: row softmax over 512 ---------------------------
__global__ __launch_bounds__(256) void softmax_kernel(const float* __restrict__ logits,
                                                      float* __restrict__ out) {
    __shared__ float red[8];
    const int b   = blockIdx.x;
    const int tid = threadIdx.x;
    const float x0 = logits[(size_t)b * SRC_LEN + tid];
    const float x1 = logits[(size_t)b * SRC_LEN + 256 + tid];

    float m = fmaxf(x0, x1);
    #pragma unroll
    for (int off = 16; off >= 1; off >>= 1) m = fmaxf(m, __shfl_xor(m, off, 32));
    if ((tid & 31) == 0) red[tid >> 5] = m;
    __syncthreads();
    float gm = red[0];
    #pragma unroll
    for (int i = 1; i < 8; ++i) gm = fmaxf(gm, red[i]);
    __syncthreads();

    const float e0 = __expf(x0 - gm);
    const float e1 = __expf(x1 - gm);
    float sum = e0 + e1;
    #pragma unroll
    for (int off = 16; off >= 1; off >>= 1) sum += __shfl_xor(sum, off, 32);
    if ((tid & 31) == 0) red[tid >> 5] = sum;
    __syncthreads();
    float gs = 0.f;
    #pragma unroll
    for (int i = 0; i < 8; ++i) gs += red[i];
    const float inv = 1.0f / gs;

    out[(size_t)b * SRC_LEN + tid]       = e0 * inv;
    out[(size_t)b * SRC_LEN + 256 + tid] = e1 * inv;
}

extern "C" void kernel_launch(void* const* d_in, const int* in_sizes, int n_in,
                              void* d_out, int out_size, void* d_ws, size_t ws_size,
                              hipStream_t stream) {
    const float* enc  = (const float*)d_in[0];  // [512,64,2048]
    const float* dec  = (const float*)d_in[1];  // [64,1024]
    const float* Ua_w = (const float*)d_in[2];  // [1024,2048]
    const float* Ua_b = (const float*)d_in[3];  // [1024]
    const float* Wa_w = (const float*)d_in[4];  // [1024,1024]
    const float* Wa_b = (const float*)d_in[5];  // [1024]
    const float* v_w  = (const float*)d_in[6];  // [1024]

    char* ws = (char*)d_ws;
    unsigned short* ua_bf  = (unsigned short*)ws;                         // 4 MB
    float*          proj   = (float*)(ws + (size_t)UNIT * HID2 * 2);      // 256 KB
    float*          logits = (float*)(ws + (size_t)UNIT * HID2 * 2
                                         + (size_t)BATCH * UNIT * 4);     // 128 KB

    const int n4 = UNIT * HID2 / 4;  // 524288 float4
    cvt_ua_kernel<<<(n4 + 255) / 256, 256, 0, stream>>>(Ua_w, (unsigned int*)ua_bf, n4);
    dec_proj_kernel<<<BATCH, 256, 0, stream>>>(dec, Wa_w, Wa_b, Ua_b, proj);
    attn_main_kernel<<<SRC_LEN, 512, 0, stream>>>(enc, ua_bf, proj, v_w, logits);
    softmax_kernel<<<BATCH, 256, 0, stream>>>(logits, (float*)d_out);
}
